// HorizonForcingGRULayer_87677462381238
// MI455X (gfx1250) — matrix operand order repaired and checked by reading, not compiled, not run on hardware
//
#include <hip/hip_runtime.h>

typedef __bf16 v16bf __attribute__((ext_vector_type(16)));
typedef __bf16 bf16x8 __attribute__((ext_vector_type(8)));
typedef __bf16 bf16x4 __attribute__((ext_vector_type(4)));
typedef float  v8f    __attribute__((ext_vector_type(8)));
typedef float  f32x4  __attribute__((ext_vector_type(4)));

static constexpr int BT = 32 * 512;   // 16384 rows (B*T)
static constexpr int Hd = 1024;       // hidden / K of all GEMMs
static constexpr int Dd = 1024;       // dense out dim

// ---------------------------------------------------------------------------
// fp32 -> bf16 bulk convert (vectorized)
// ---------------------------------------------------------------------------
__global__ __launch_bounds__(256) void conv_bf16_kernel(
    const float* __restrict__ in, __bf16* __restrict__ out, int n) {
  int i = (blockIdx.x * 256 + threadIdx.x) * 4;
  if (i < n) {
    f32x4 v = *(const f32x4*)(in + i);
    bf16x4 o;
    o[0] = (__bf16)v[0]; o[1] = (__bf16)v[1];
    o[2] = (__bf16)v[2]; o[3] = (__bf16)v[3];
    *(bf16x4*)(out + i) = o;
  }
}

// ---------------------------------------------------------------------------
// fp32 [K][N] row-major -> bf16 [N][K] (transpose + convert). Reads coalesced.
// ---------------------------------------------------------------------------
__global__ __launch_bounds__(256) void transpose_bf16_kernel(
    const float* __restrict__ in, __bf16* __restrict__ out, int K, int N) {
  int tid = blockIdx.x * 256 + threadIdx.x;
  if (tid < K * N) {
    int k = tid / N;
    int n = tid - k * N;
    out[n * K + k] = (__bf16)in[tid];
  }
}

// ---------------------------------------------------------------------------
// WMMA fragment loaders (wave32, 16x16x32 bf16 shapes)
// A 16x32 (MxK), row-major, leading dim ld:
//   lanes 0-15:  M=lane,    K chunks [0..7] and [16..23]
//   lanes 16-31: M=lane-16, K chunks [8..15] and [24..31]
// ---------------------------------------------------------------------------
__device__ __forceinline__ v16bf load_frag_a(const __bf16* base, int ld, int lane) {
  int m  = lane & 15;
  int kb = (lane < 16) ? 0 : 8;
  const __bf16* p = base + m * ld + kb;
  bf16x8 lo = *(const bf16x8*)(p);
  bf16x8 hi = *(const bf16x8*)(p + 16);
  return __builtin_shufflevector(lo, hi, 0,1,2,3,4,5,6,7,8,9,10,11,12,13,14,15);
}

// B 32x16 (KxN) supplied from a transposed weight buffer Bt[N][K], lead dim ld:
//   lanes 0-15:  N=lane,    K chunks [0..7] and [16..23]
//   lanes 16-31: N=lane-16, K chunks [8..15] and [24..31]
__device__ __forceinline__ v16bf load_frag_b(const __bf16* base, int ld, int lane) {
  int n  = lane & 15;
  int kb = (lane < 16) ? 0 : 8;
  const __bf16* p = base + n * ld + kb;
  bf16x8 lo = *(const bf16x8*)(p);
  bf16x8 hi = *(const bf16x8*)(p + 16);
  return __builtin_shufflevector(lo, hi, 0,1,2,3,4,5,6,7,8,9,10,11,12,13,14,15);
}

#define WMMA_BF16(A, B, C) \
  __builtin_amdgcn_wmma_f32_16x16x32_bf16(false, (A), false, (B), (short)0, (C), false, false)

// ---------------------------------------------------------------------------
// Kernel 1: pred = hb @ W_dense + b_dense
// block = 256 thr = 8 waves, tile 128(M) x 64(N); wave tile 32x32
// fp32 pred -> d_out via NT stores (write-only stream, keep it out of L2);
// bf16 pred copy -> ws via RT stores (re-read 32x by kernel 2, keep in L2).
// ---------------------------------------------------------------------------
__global__ __launch_bounds__(256) void gemm_dense_kernel(
    const __bf16* __restrict__ hb, const __bf16* __restrict__ WdT,
    const float* __restrict__ b_dense,
    float* __restrict__ pred, __bf16* __restrict__ predb) {
  const int lane = threadIdx.x & 31;
  const int w    = threadIdx.x >> 5;
  const int wm   = w & 3;          // 4 waves in M
  const int wn   = w >> 2;         // 2 waves in N
  const int Rm   = blockIdx.x * 128 + wm * 32;
  const int Cn   = blockIdx.y * 64  + wn * 32;

  v8f acc00 = {}, acc01 = {}, acc10 = {}, acc11 = {};
  const __bf16* Ab = hb  + Rm * Hd;
  const __bf16* Bb = WdT + Cn * Hd;

  for (int kk = 0; kk < Hd; kk += 32) {
    v16bf a0 = load_frag_a(Ab + kk,           Hd, lane);
    v16bf a1 = load_frag_a(Ab + 16 * Hd + kk, Hd, lane);
    v16bf b0 = load_frag_b(Bb + kk,           Hd, lane);
    v16bf b1 = load_frag_b(Bb + 16 * Hd + kk, Hd, lane);
    acc00 = WMMA_BF16(a0, b0, acc00);
    acc01 = WMMA_BF16(a0, b1, acc01);
    acc10 = WMMA_BF16(a1, b0, acc10);
    acc11 = WMMA_BF16(a1, b1, acc11);
  }

  // epilogue: C/D layout -> VGPR r holds (M = r + (lane>=16?8:0), N = lane&15)
  const int col   = lane & 15;
  const int rbase = (lane < 16) ? 0 : 8;
  v8f accs[2][2] = {{acc00, acc01}, {acc10, acc11}};
#pragma unroll
  for (int mt = 0; mt < 2; ++mt) {
#pragma unroll
    for (int nt = 0; nt < 2; ++nt) {
      int c0 = Cn + nt * 16 + col;
      float bias = b_dense[c0];
#pragma unroll
      for (int r = 0; r < 8; ++r) {
        int row = Rm + mt * 16 + rbase + r;
        float v = accs[mt][nt][r] + bias;
        __builtin_nontemporal_store(v, pred + row * Dd + c0);  // write-only stream
        predb[row * Dd + c0] = (__bf16)v;                      // hot in L2 for K2
      }
    }
  }
}

// ---------------------------------------------------------------------------
// Kernel 2: fused 6-way GEMM + GRU gates (reset_after=True)
//   xz/xr/xh = predb @ W_gru[:, {0,H,2H}+n] ; uz/ur/uh = hb @ U_gru[...]
//   z=sig(xz+uz); r=sig(xr+ur); hh=tanh(xh + r*uh); h_new = z*h + (1-z)*hh
// block = 256 thr = 8 waves, tile 128(M) x 32(N); wave tile 32x16
// 12 WMMAs / k-step / wave; A-fragments shared across the 3 gate columns.
// ---------------------------------------------------------------------------
__global__ __launch_bounds__(256) void gru_fused_kernel(
    const __bf16* __restrict__ predb, const __bf16* __restrict__ hb,
    const __bf16* __restrict__ WgT,   const __bf16* __restrict__ UgT,
    const float* __restrict__ b_gru,  const float* __restrict__ h_in,
    float* __restrict__ h_out) {
  const int lane = threadIdx.x & 31;
  const int w    = threadIdx.x >> 5;
  const int wm   = w & 3;          // 4 waves in M
  const int wn   = w >> 2;         // 2 waves in N
  const int Rm   = blockIdx.x * 128 + wm * 32;
  const int Cn   = blockIdx.y * 32  + wn * 16;

  v8f axz0 = {}, axz1 = {}, axr0 = {}, axr1 = {}, axh0 = {}, axh1 = {};
  v8f auz0 = {}, auz1 = {}, aur0 = {}, aur1 = {}, auh0 = {}, auh1 = {};

  const __bf16* Ap = predb + Rm * Dd;   // A for x-gates
  const __bf16* Ah = hb    + Rm * Hd;   // A for u-gates
  const __bf16* Bz = WgT + (Cn)        * Dd;
  const __bf16* Br = WgT + (Cn + 1024) * Dd;
  const __bf16* Bh = WgT + (Cn + 2048) * Dd;
  const __bf16* Cz = UgT + (Cn)        * Hd;
  const __bf16* Cr = UgT + (Cn + 1024) * Hd;
  const __bf16* Ch = UgT + (Cn + 2048) * Hd;

  for (int kk = 0; kk < Hd; kk += 32) {
    v16bf ap0 = load_frag_a(Ap + kk,           Dd, lane);
    v16bf ap1 = load_frag_a(Ap + 16 * Dd + kk, Dd, lane);
    v16bf ah0 = load_frag_a(Ah + kk,           Hd, lane);
    v16bf ah1 = load_frag_a(Ah + 16 * Hd + kk, Hd, lane);
    v16bf bz = load_frag_b(Bz + kk, Dd, lane);
    v16bf br = load_frag_b(Br + kk, Dd, lane);
    v16bf bh = load_frag_b(Bh + kk, Dd, lane);
    v16bf cz = load_frag_b(Cz + kk, Hd, lane);
    v16bf cr = load_frag_b(Cr + kk, Hd, lane);
    v16bf ch = load_frag_b(Ch + kk, Hd, lane);
    axz0 = WMMA_BF16(ap0, bz, axz0);  axz1 = WMMA_BF16(ap1, bz, axz1);
    axr0 = WMMA_BF16(ap0, br, axr0);  axr1 = WMMA_BF16(ap1, br, axr1);
    axh0 = WMMA_BF16(ap0, bh, axh0);  axh1 = WMMA_BF16(ap1, bh, axh1);
    auz0 = WMMA_BF16(ah0, cz, auz0);  auz1 = WMMA_BF16(ah1, cz, auz1);
    aur0 = WMMA_BF16(ah0, cr, aur0);  aur1 = WMMA_BF16(ah1, cr, aur1);
    auh0 = WMMA_BF16(ah0, ch, auh0);  auh1 = WMMA_BF16(ah1, ch, auh1);
  }

  const int col   = Cn + (lane & 15);
  const int rbase = (lane < 16) ? 0 : 8;
  const float bxz = b_gru[col],        bxr = b_gru[1024 + col], bxh = b_gru[2048 + col];
  const float buz = b_gru[3072 + col], bur = b_gru[4096 + col], buh = b_gru[5120 + col];

  v8f XZ[2] = {axz0, axz1}, XR[2] = {axr0, axr1}, XH[2] = {axh0, axh1};
  v8f UZ[2] = {auz0, auz1}, UR[2] = {aur0, aur1}, UH[2] = {auh0, auh1};

#pragma unroll
  for (int mt = 0; mt < 2; ++mt) {
#pragma unroll
    for (int r = 0; r < 8; ++r) {
      int row = Rm + mt * 16 + rbase + r;
      float xz = XZ[mt][r] + bxz, xr = XR[mt][r] + bxr, xh = XH[mt][r] + bxh;
      float uz = UZ[mt][r] + buz, ur = UR[mt][r] + bur, uh = UH[mt][r] + buh;
      float z  = 1.0f / (1.0f + __expf(-(xz + uz)));
      float rg = 1.0f / (1.0f + __expf(-(xr + ur)));
      float hh = tanhf(xh + rg * uh);
      float h  = __builtin_nontemporal_load(h_in + row * Hd + col);  // last use
      float hn = z * h + (1.0f - z) * hh;
      __builtin_nontemporal_store(hn, h_out + row * Hd + col);       // write-only
    }
  }
}

// ---------------------------------------------------------------------------
extern "C" void kernel_launch(void* const* d_in, const int* in_sizes, int n_in,
                              void* d_out, int out_size, void* d_ws, size_t ws_size,
                              hipStream_t stream) {
  const float* inputs  = (const float*)d_in[0];   // [BT, H]
  const float* W_dense = (const float*)d_in[1];   // [H, D]
  const float* b_dense = (const float*)d_in[2];   // [D]
  const float* W_gru   = (const float*)d_in[3];   // [D, 3H]
  const float* U_gru   = (const float*)d_in[4];   // [H, 3H]
  const float* b_gru   = (const float*)d_in[5];   // [2, 3H]

  float* pred  = (float*)d_out;                   // [BT, D]
  float* h_new = pred + (size_t)BT * Dd;          // [BT, H]

  // workspace layout (bytes)
  char* ws = (char*)d_ws;
  __bf16* hb    = (__bf16*)(ws);                    // BT*H bf16   = 32 MB
  __bf16* predb = (__bf16*)(ws + 33554432);         // BT*D bf16   = 32 MB
  __bf16* WdT   = (__bf16*)(ws + 67108864);         // [D][H]      =  2 MB
  __bf16* WgT   = (__bf16*)(ws + 69206016);         // [3H][D]     =  6 MB
  __bf16* UgT   = (__bf16*)(ws + 75497472);         // [3H][H]     =  6 MB

  // 0) precision conversion + weight transposes
  {
    int n = BT * Hd;                                // 16,777,216
    conv_bf16_kernel<<<n / (256 * 4), 256, 0, stream>>>(inputs, hb, n);
  }
  transpose_bf16_kernel<<<(Hd * Dd + 255) / 256, 256, 0, stream>>>(W_dense, WdT, Hd, Dd);
  transpose_bf16_kernel<<<(Dd * 3072 + 255) / 256, 256, 0, stream>>>(W_gru, WgT, Dd, 3072);
  transpose_bf16_kernel<<<(Hd * 3072 + 255) / 256, 256, 0, stream>>>(U_gru, UgT, Hd, 3072);

  // 1) pred = hb @ W_dense + b_dense   (fp32 NT out + bf16 copy)
  dim3 g1(BT / 128, Dd / 64);
  gemm_dense_kernel<<<g1, 256, 0, stream>>>(hb, WdT, b_dense, pred, predb);

  // 2) fused 6-GEMM + GRU gate math -> h_new
  dim3 g2(BT / 128, Hd / 32);
  gru_fused_kernel<<<g2, 256, 0, stream>>>(predb, hb, WgT, UgT, b_gru, inputs, h_new);
}
